// GraphModel_74998718923362
// MI455X (gfx1250) — matrix operand order repaired and checked by reading, not compiled, not run on hardware
//
#include <hip/hip_runtime.h>
#include <hip/hip_bf16.h>

// GIN layer for MI455X (gfx1250, wave32, WMMA).
// Edge message matmul (K=96 incl. one-hot block) + node MLP with
// v_wmma_f32_16x16x32_bf16; scatter via coalesced f32 atomics into
// an L2-resident aggregation table.

typedef __attribute__((ext_vector_type(16))) __bf16 v16bf;
typedef __attribute__((ext_vector_type(8)))  float  v8f;

#define NN   100000
#define NE   3200000
#define D    64
#define FAN  96
#define NET  32
#define BN_EPS 1e-5f

// float -> bf16: native fptrunc (RNE). gfx1250 has first-class BF16 VALU,
// so this should lower to v_cvt_pk_bf16_f32-class instructions instead of
// the 3-4 integer-op manual rounding sequence.
__device__ __forceinline__ __bf16 f2bf(float f) {
  return (__bf16)f;
}

__device__ __forceinline__ v8f wmma_bf16(v16bf a, v16bf b, v8f c) {
  return __builtin_amdgcn_wmma_f32_16x16x32_bf16(
      /*neg_a=*/false, a, /*neg_b=*/false, b,
      /*c_mod=*/(short)0, c, /*reuse_a=*/false, /*reuse_b=*/false);
}

// A-matrix (16x32 bf16) per-lane loader, CDNA5 layout:
// lanes 0-15 (khalf=0): elems 0..7 -> K = kbase+0..7,  elems 8..15 -> K = kbase+16..23
// lanes 16-31 (khalf=1): +8 on both ranges.
__device__ __forceinline__ v16bf loadA(const float* __restrict__ row, int khalf, int kbase) {
  const float4* p0 = (const float4*)(row + kbase + 8 * khalf);
  const float4* p1 = (const float4*)(row + kbase + 16 + 8 * khalf);
  float4 a = p0[0], b = p0[1], c = p1[0], d = p1[1];
  v16bf r;
  r[0]=f2bf(a.x); r[1]=f2bf(a.y); r[2]=f2bf(a.z); r[3]=f2bf(a.w);
  r[4]=f2bf(b.x); r[5]=f2bf(b.y); r[6]=f2bf(b.z); r[7]=f2bf(b.w);
  r[8]=f2bf(c.x); r[9]=f2bf(c.y); r[10]=f2bf(c.z); r[11]=f2bf(c.w);
  r[12]=f2bf(d.x); r[13]=f2bf(d.y); r[14]=f2bf(d.z); r[15]=f2bf(d.w);
  return r;
}

// Same A layout, fusing h = eps1*x + agg.
__device__ __forceinline__ v16bf loadA_h(const float* __restrict__ xr,
                                         const float* __restrict__ ar,
                                         float eps1, int khalf, int kbase) {
  int k0 = kbase + 8 * khalf;
  int k1 = kbase + 16 + 8 * khalf;
  v16bf r;
#pragma unroll
  for (int i = 0; i < 8; ++i) {
    r[i]     = f2bf(eps1 * xr[k0 + i] + ar[k0 + i]);
    r[8 + i] = f2bf(eps1 * xr[k1 + i] + ar[k1 + i]);
  }
  return r;
}

// Same A layout, fusing BN affine + relu: a = max(h1*scale + shift, 0).
__device__ __forceinline__ v16bf loadA_act(const float* __restrict__ hr,
                                           const float* __restrict__ sc,
                                           const float* __restrict__ sh,
                                           int khalf, int kbase) {
  int k0 = kbase + 8 * khalf;
  int k1 = kbase + 16 + 8 * khalf;
  v16bf r;
#pragma unroll
  for (int i = 0; i < 8; ++i) {
    r[i]     = f2bf(fmaxf(hr[k0 + i] * sc[k0 + i] + sh[k0 + i], 0.0f));
    r[8 + i] = f2bf(fmaxf(hr[k1 + i] * sc[k1 + i] + sh[k1 + i], 0.0f));
  }
  return r;
}

// One-hot A tile for the edge-type block (K local 0..31), built in registers.
// A-layout local K of element e: kl = e + (e>=8 ? 8 : 0) + 8*khalf.
__device__ __forceinline__ v16bf onehotA(int et, int khalf) {
  const __bf16 one  = __builtin_bit_cast(__bf16, (unsigned short)0x3F80);
  const __bf16 zero = __builtin_bit_cast(__bf16, (unsigned short)0x0000);
  v16bf r;
#pragma unroll
  for (int e = 0; e < 16; ++e) {
    int kl = e + ((e >= 8) ? 8 : 0) + 8 * khalf;
    r[e] = (kl == et) ? one : zero;
  }
  return r;
}

// B-matrix (32x16 bf16) per-lane loader: lane holds column n, same K layout as A:
// elems 0..7 -> K kbase+8*khalf+0..7, elems 8..15 -> K kbase+16+8*khalf+0..7.
__device__ __forceinline__ v16bf loadB(const float* __restrict__ row, int khalf, int kbase) {
  const float* p0 = row + kbase + 8 * khalf;
  const float* p1 = row + kbase + 16 + 8 * khalf;
  v16bf r;
#pragma unroll
  for (int i = 0; i < 8; ++i) { r[i] = f2bf(p0[i]); r[8 + i] = f2bf(p1[i]); }
  return r;
}

// ---------------------------------------------------------------------------
// Kernel 1: edge messages (K=96 WMMA incl. one-hot block) + scatter-add.
// ---------------------------------------------------------------------------
__global__ void __launch_bounds__(256)
edge_kernel(const float* __restrict__ x, const float* __restrict__ Wedge,
            const float* __restrict__ bedge, const int* __restrict__ src,
            const int* __restrict__ tgt, const int* __restrict__ etyp,
            float* __restrict__ agg) {
  const int tid = threadIdx.x;
  const int lane = tid & 31;
  const int wid = tid >> 5;
  const int m = lane & 15;
  const int khalf = lane >> 4;

  // Preload B tiles of W_edge^T (3 K-blocks: x part 0..63, one-hot part 64..95)
  // and the per-column bias b_edge[n] for each N tile.
  v16bf Bt[4][3];
  float bb[4];
#pragma unroll
  for (int nt = 0; nt < 4; ++nt) {
    const float* wr = Wedge + (size_t)(m + 16 * nt) * FAN;
    Bt[nt][0] = loadB(wr, khalf, 0);
    Bt[nt][1] = loadB(wr, khalf, 32);
    Bt[nt][2] = loadB(wr, khalf, 64);
    bb[nt] = bedge[m + 16 * nt];
  }

  const int tiles = NE / 16;  // 200000, exact
  const int gw = blockIdx.x * (blockDim.x >> 5) + wid;
  const int nw = gridDim.x * (blockDim.x >> 5);

  for (int t = gw; t < tiles; t += nw) {
    const int e = t * 16 + m;          // lanes 0-15 / 16-31 duplicate edges 0..15
    const int s = src[e];
    const int g = tgt[e];
    int et = etyp[e];
    et = et < 0 ? 0 : (et > NET - 1 ? NET - 1 : et);
    const int packed = (g << 1) | (int)(s != g);   // tgt + valid in one shuffle

    const float* xr = x + (size_t)s * D;           // gather (L2-resident table)
    v16bf a0 = loadA(xr, khalf, 0);
    v16bf a1 = loadA(xr, khalf, 32);
    v16bf a2 = onehotA(et, khalf);

    // Per-C-row target pointers + validity, hoisted out of the N-tile loop.
    float* pr[8];
    float vmf[8];
#pragma unroll
    for (int r = 0; r < 8; ++r) {
      const int pk = __shfl(packed, r + 8 * khalf, 32);
      vmf[r] = (float)(pk & 1);
      pr[r] = agg + (size_t)(pk >> 1) * D + m;
    }

#pragma unroll
    for (int nt = 0; nt < 4; ++nt) {
      v8f c = {};
      c = wmma_bf16(a0, Bt[nt][0], c);
      c = wmma_bf16(a1, Bt[nt][1], c);
      c = wmma_bf16(a2, Bt[nt][2], c);
#pragma unroll
      for (int r = 0; r < 8; ++r) {
        // relu(msg+bias); invalid (self-loop) edges contribute exactly 0.0.
        float v = fmaxf(c[r] + bb[nt], 0.0f) * vmf[r];
        atomicAdd(pr[r] + nt * 16, v);   // nt*16 folds into ioffset; lanes
      }                                  // 0-15/16-31 hit 2 contiguous rows
    }
  }
}

// ---------------------------------------------------------------------------
// Kernel 2: h = (1+eps)x + agg ; h1 = h @ W1^T ; accumulate column sums/sumsq.
// ---------------------------------------------------------------------------
__global__ void __launch_bounds__(256)
node_h1_kernel(const float* __restrict__ x, const float* __restrict__ agg,
               const float* __restrict__ W1, const float* __restrict__ epsp,
               float* __restrict__ h1, float* __restrict__ sumb,
               float* __restrict__ sqb) {
  const int tid = threadIdx.x;
  const int lane = tid & 31, wid = tid >> 5;
  const int m = lane & 15, khalf = lane >> 4;
  const float eps1 = 1.0f + epsp[0];

  v16bf Bt[4][2];
#pragma unroll
  for (int nt = 0; nt < 4; ++nt) {
    const float* wr = W1 + (size_t)(m + 16 * nt) * D;
    Bt[nt][0] = loadB(wr, khalf, 0);
    Bt[nt][1] = loadB(wr, khalf, 32);
  }

  const int tiles = NN / 16;  // 6250, exact
  const int gw = blockIdx.x * (blockDim.x >> 5) + wid;
  const int nw = gridDim.x * (blockDim.x >> 5);

  for (int t = gw; t < tiles; t += nw) {
    const int node = t * 16 + m;
    const float* xr = x + (size_t)node * D;
    const float* ar = agg + (size_t)node * D;
    v16bf a0 = loadA_h(xr, ar, eps1, khalf, 0);
    v16bf a1 = loadA_h(xr, ar, eps1, khalf, 32);

#pragma unroll
    for (int nt = 0; nt < 4; ++nt) {
      v8f c = {};
      c = wmma_bf16(a0, Bt[nt][0], c);
      c = wmma_bf16(a1, Bt[nt][1], c);
      const int n = m + 16 * nt;
      float s = 0.0f, q = 0.0f;
#pragma unroll
      for (int r = 0; r < 8; ++r) {
        const int row = t * 16 + 8 * khalf + r;
        h1[(size_t)row * D + n] = c[r];
        s += c[r];
        q += c[r] * c[r];
      }
      atomicAdd(sumb + n, s);
      atomicAdd(sqb + n, q);
    }
  }
}

// ---------------------------------------------------------------------------
// Kernel 3: fold batch stats into affine scale/shift (population variance).
// ---------------------------------------------------------------------------
__global__ void stats_kernel(const float* __restrict__ sumb,
                             const float* __restrict__ sqb,
                             const float* __restrict__ gamma,
                             const float* __restrict__ beta,
                             float* __restrict__ scale,
                             float* __restrict__ shift) {
  const int n = threadIdx.x;
  const float invN = 1.0f / (float)NN;
  float mu = sumb[n] * invN;
  float var = sqb[n] * invN - mu * mu;
  float rs = rsqrtf(var + BN_EPS);
  float sc = rs * gamma[n];
  scale[n] = sc;
  shift[n] = beta[n] - mu * sc;
}

// ---------------------------------------------------------------------------
// Kernel 4: out = relu(h1*scale + shift) @ W2^T.
// ---------------------------------------------------------------------------
__global__ void __launch_bounds__(256)
node_out_kernel(const float* __restrict__ h1, const float* __restrict__ W2,
                const float* __restrict__ scale, const float* __restrict__ shift,
                float* __restrict__ out) {
  __shared__ float s_sc[D], s_sh[D];
  const int tid = threadIdx.x;
  if (tid < D) { s_sc[tid] = scale[tid]; s_sh[tid] = shift[tid]; }
  __syncthreads();

  const int lane = tid & 31, wid = tid >> 5;
  const int m = lane & 15, khalf = lane >> 4;

  v16bf Bt[4][2];
#pragma unroll
  for (int nt = 0; nt < 4; ++nt) {
    const float* wr = W2 + (size_t)(m + 16 * nt) * D;
    Bt[nt][0] = loadB(wr, khalf, 0);
    Bt[nt][1] = loadB(wr, khalf, 32);
  }

  const int tiles = NN / 16;
  const int gw = blockIdx.x * (blockDim.x >> 5) + wid;
  const int nw = gridDim.x * (blockDim.x >> 5);

  for (int t = gw; t < tiles; t += nw) {
    const int node = t * 16 + m;
    const float* hr = h1 + (size_t)node * D;
    v16bf a0 = loadA_act(hr, s_sc, s_sh, khalf, 0);
    v16bf a1 = loadA_act(hr, s_sc, s_sh, khalf, 32);

#pragma unroll
    for (int nt = 0; nt < 4; ++nt) {
      v8f c = {};
      c = wmma_bf16(a0, Bt[nt][0], c);
      c = wmma_bf16(a1, Bt[nt][1], c);
      const int n = m + 16 * nt;
#pragma unroll
      for (int r = 0; r < 8; ++r) {
        const int row = t * 16 + 8 * khalf + r;
        out[(size_t)row * D + n] = c[r];
      }
    }
  }
}

// ---------------------------------------------------------------------------
extern "C" void kernel_launch(void* const* d_in, const int* in_sizes, int n_in,
                              void* d_out, int out_size, void* d_ws, size_t ws_size,
                              hipStream_t stream) {
  const float* x     = (const float*)d_in[0];   // node_features [100000,64]
  const float* Wedge = (const float*)d_in[1];   // [64,96]
  const float* bedge = (const float*)d_in[2];   // [64]
  const float* W1    = (const float*)d_in[3];   // [64,64]
  const float* W2    = (const float*)d_in[4];   // [64,64]
  const float* gamma = (const float*)d_in[5];   // [64]
  const float* beta  = (const float*)d_in[6];   // [64]
  const float* eps   = (const float*)d_in[7];   // scalar
  const int* asrc    = (const int*)d_in[8];     // [3.2M]
  const int* atgt    = (const int*)d_in[9];     // [3.2M]
  const int* etyp    = (const int*)d_in[10];    // [3.2M]
  float* out = (float*)d_out;

  // Workspace layout: agg | sum | sumsq | scale | shift | h1
  float* agg   = (float*)d_ws;                  // NN*D floats (25.6 MB)
  float* sumb  = agg + (size_t)NN * D;          // 64
  float* sqb   = sumb + D;                      // 64
  float* scl   = sqb + D;                       // 64
  float* shf   = scl + D;                       // 64
  float* h1    = shf + D;                       // NN*D floats (25.6 MB)

  // Zero agg + sum + sumsq in one contiguous capturable memset.
  hipMemsetAsync(agg, 0, ((size_t)NN * D + 2 * D) * sizeof(float), stream);

  edge_kernel<<<1024, 256, 0, stream>>>(x, Wedge, bedge, asrc, atgt, etyp, agg);
  node_h1_kernel<<<256, 256, 0, stream>>>(x, agg, W1, eps, h1, sumb, sqb);
  stats_kernel<<<1, 64, 0, stream>>>(sumb, sqb, gamma, beta, scl, shf);
  node_out_kernel<<<256, 256, 0, stream>>>(h1, W2, scl, shf, out);
}
